// GATv2Backbone_50654844289321
// MI455X (gfx1250) — compile-verified, matrix operand order checked by reading, not run on hardware
//
#include <hip/hip_runtime.h>
#include <hip/hip_bf16.h>

typedef __bf16 bf16_t;
typedef __attribute__((ext_vector_type(16))) __bf16 v16bf;
typedef __attribute__((ext_vector_type(8)))  __bf16 v8bf;
typedef __attribute__((ext_vector_type(8)))  float  v8f;

#define HEADS 4

// ---------------------------------------------------------------------------
// float -> bf16 elementwise
__global__ void gat_f32_to_bf16(const float* __restrict__ s, bf16_t* __restrict__ d, int n) {
  int i = blockIdx.x * blockDim.x + threadIdx.x;
  if (i < n) d[i] = (bf16_t)s[i];
}

// W [K, NO] f32 row-major  ->  Wt [NO, K] bf16 (transposed for per-lane K-contiguous B frags)
__global__ void gat_convert_w(const float* __restrict__ W, bf16_t* __restrict__ Wt, int K, int NO) {
  int i = blockIdx.x * blockDim.x + threadIdx.x;
  if (i >= K * NO) return;
  int k = i / NO, n = i % NO;
  Wt[(size_t)n * K + k] = (bf16_t)W[i];
}

// ---------------------------------------------------------------------------
// C[M,NOUT] = A[M,128](bf16) @ Bt[NOUT,128]^T(bf16) + bias  (f32 accum via WMMA)
// one wave -> one 16(M) x 64(N) tile: A fragment loaded once per K-step,
// reused by 4 v_wmma_f32_16x16x32_bf16 against 4 B fragments (4x A reuse).
__global__ void __launch_bounds__(256)
gat_gemm_bf16_wmma(const bf16_t* __restrict__ A, const bf16_t* __restrict__ Bt,
                   const float* __restrict__ bias, float* __restrict__ C,
                   int M, int NOUT) {
  const int lane = threadIdx.x & 31;
  const int wave = threadIdx.x >> 5;
  const int ntN  = NOUT >> 6;                 // 64-wide N tiles
  const int ntiles = (M >> 4) * ntN;
  int tile = blockIdx.x * 8 + wave;
  if (tile >= ntiles) return;
  const int tM = tile / ntN, tN = tile % ntN;
  const int hf = lane >> 4;                   // K sub-block selector
  const int q  = lane & 15;                   // M index (A) / N index (B,C)

  const bf16_t* arow  = A  + (size_t)(tM * 16 + q) * 128;
  const bf16_t* bbase = Bt + (size_t)(tN * 64 + q) * 128;   // cols q, q+16, q+32, q+48

  v8f acc[4] = {{}, {}, {}, {}};
#pragma unroll
  for (int kk = 0; kk < 128; kk += 32) {
    if (kk < 96) {  // WGP-scope global_prefetch_b8 for next K block
      __builtin_prefetch((const void*)(arow + kk + 32), 0, 3);
      __builtin_prefetch((const void*)(bbase + kk + 32), 0, 3);
    }
    // 16-bit A 16x32 layout: lanes 0-15 hold K {0..7,16..23}, lanes 16-31 hold K {8..15,24..31}
    v8bf alo = *(const v8bf*)(arow + kk + 8 * hf);
    v8bf ahi = *(const v8bf*)(arow + kk + 16 + 8 * hf);
    v16bf a;
#pragma unroll
    for (int i = 0; i < 8; ++i) { a[i] = alo[i]; a[i + 8] = ahi[i]; }
#pragma unroll
    for (int j = 0; j < 4; ++j) {
      const bf16_t* bcol = bbase + (size_t)j * 16 * 128;
      v8bf blo = *(const v8bf*)(bcol + kk + 8 * hf);
      v8bf bhi = *(const v8bf*)(bcol + kk + 16 + 8 * hf);
      v16bf b;
#pragma unroll
      for (int i = 0; i < 8; ++i) { b[i] = blo[i]; b[i + 8] = bhi[i]; }
      acc[j] = __builtin_amdgcn_wmma_f32_16x16x32_bf16(false, a, false, b, (short)0, acc[j],
                                                       false, false);
    }
  }

#pragma unroll
  for (int j = 0; j < 4; ++j) {
    const int col = tN * 64 + j * 16 + q;
    const float bv = bias[col];
    float* crow = C + (size_t)(tM * 16 + 8 * hf) * NOUT + col;  // C: VGPR v -> row v + 8*hf
#pragma unroll
    for (int v = 0; v < 8; ++v) crow[(size_t)v * NOUT] = acc[j][v] + bv;
  }
}

// ---------------------------------------------------------------------------
__global__ void gat_init_bufs(float* __restrict__ maxv, float* __restrict__ denom,
                              float* __restrict__ acc, int nmh, int nacc) {
  int i = blockIdx.x * blockDim.x + threadIdx.x;
  if (i < nmh) { maxv[i] = -3.0e38f; denom[i] = 0.f; }
  if (i < nacc) acc[i] = 0.f;
}

__device__ inline void atomicMaxF(float* addr, float value) {
  if (value >= 0.f) atomicMax((int*)addr, __float_as_int(value));
  else              atomicMin((unsigned int*)addr, (unsigned int)__float_as_int(value));
}

// wave per edge: logits[e,h] = sum_c leakyrelu(xl[src,h,c]+xr[dst,h,c]) * att[h,c]; segment-max into maxv
__global__ void __launch_bounds__(256)
gat_edge_logits_max(const int* __restrict__ esrc, const int* __restrict__ edst,
                    int nE, int nNodes,
                    const float* __restrict__ xl, const float* __restrict__ xr,
                    const float* __restrict__ att,
                    float* __restrict__ logits, float* __restrict__ maxv, int C) {
  const int lane = threadIdx.x & 31;
  const int e = blockIdx.x * 8 + (threadIdx.x >> 5);
  if (e >= nE + nNodes) return;
  int s, d;
  if (e < nE) { s = esrc[e]; d = edst[e]; } else { s = e - nE; d = s; }
  const float* pl = xl + (size_t)s * HEADS * C;
  const float* pr = xr + (size_t)d * HEADS * C;
#pragma unroll
  for (int hh = 0; hh < HEADS; ++hh) {
    float a = 0.f;
    for (int j = lane; j < C; j += 32) {
      float v = pl[hh * C + j] + pr[hh * C + j];
      v = v > 0.f ? v : 0.2f * v;           // GATv2 leaky_relu(0.2) before dot
      a += v * att[hh * C + j];
    }
    a += __shfl_xor(a, 16); a += __shfl_xor(a, 8); a += __shfl_xor(a, 4);
    a += __shfl_xor(a, 2);  a += __shfl_xor(a, 1);
    if (lane == 0) {
      logits[(size_t)e * HEADS + hh] = a;
      atomicMaxF(&maxv[(size_t)d * HEADS + hh], a);
    }
  }
}

// thread per (edge, head): p = exp(logit - max[dst]); denom[dst] += p
__global__ void gat_edge_exp_sum(const int* __restrict__ edst, int nE, int nNodes,
                                 float* __restrict__ logits, const float* __restrict__ maxv,
                                 float* __restrict__ denom) {
  int idx = blockIdx.x * blockDim.x + threadIdx.x;
  if (idx >= (nE + nNodes) * HEADS) return;
  int e = idx >> 2, hh = idx & 3;
  int d = (e < nE) ? edst[e] : (e - nE);
  float p = __expf(logits[idx] - maxv[d * HEADS + hh]);
  logits[idx] = p;
  atomicAdd(&denom[d * HEADS + hh], p);
}

// wave per edge: out[dst] += alpha * xl[src]  (concat) or mean over heads (layer 2)
__global__ void __launch_bounds__(256)
gat_edge_scatter(const int* __restrict__ esrc, const int* __restrict__ edst,
                 int nE, int nNodes,
                 const float* __restrict__ xl, const float* __restrict__ logits,
                 const float* __restrict__ denom, float* __restrict__ outacc,
                 int C, int meanHeads) {
  const int lane = threadIdx.x & 31;
  const int e = blockIdx.x * 8 + (threadIdx.x >> 5);
  if (e >= nE + nNodes) return;
  int s, d;
  if (e < nE) { s = esrc[e]; d = edst[e]; } else { s = e - nE; d = s; }
  float alpha[HEADS];
#pragma unroll
  for (int hh = 0; hh < HEADS; ++hh)
    alpha[hh] = logits[(size_t)e * HEADS + hh] / denom[(size_t)d * HEADS + hh];
  const float* ps = xl + (size_t)s * HEADS * C;
  float* po = outacc + (size_t)d * 128;
  if (!meanHeads) {            // C == 32, concat heads -> 128 channels
#pragma unroll
    for (int hh = 0; hh < HEADS; ++hh)
      atomicAdd(&po[hh * 32 + lane], alpha[hh] * ps[hh * 32 + lane]);
  } else {                     // C == 128, mean over heads folded into scatter
#pragma unroll
    for (int j = 0; j < 4; ++j) {
      int c = lane + 32 * j;
      float a = 0.f;
#pragma unroll
      for (int hh = 0; hh < HEADS; ++hh) a += alpha[hh] * ps[hh * 128 + c];
      atomicAdd(&po[c], 0.25f * a);
    }
  }
}

// bias + (optional ELU); write f32 final or bf16 next-layer input
__global__ void gat_epilogue(const float* __restrict__ acc, const float* __restrict__ bias,
                             float* __restrict__ outf32, bf16_t* __restrict__ hbf,
                             int total, int applyElu) {
  int i = blockIdx.x * blockDim.x + threadIdx.x;
  if (i >= total) return;
  float v = acc[i] + bias[i & 127];
  if (applyElu) v = v > 0.f ? v : (__expf(v) - 1.f);
  if (outf32) outf32[i] = v;
  if (hbf)    hbf[i] = (bf16_t)v;
}

// ---------------------------------------------------------------------------
extern "C" void kernel_launch(void* const* d_in, const int* in_sizes, int n_in,
                              void* d_out, int out_size, void* d_ws, size_t ws_size,
                              hipStream_t stream) {
  (void)n_in; (void)out_size; (void)ws_size;
  const float* x  = (const float*)d_in[0];
  const int*   ei = (const int*)d_in[1];
  const int N    = in_sizes[0] / 128;
  const int E    = in_sizes[1] / 2;
  const int Etot = E + N;                  // + self-loops
  const int* esrc = ei;
  const int* edst = ei + E;

  auto AL = [](size_t v) { return (v + 255) & ~(size_t)255; };
  char* p = (char*)d_ws;
  float*  xl     = (float*)p;  p += AL((size_t)N * 512 * 4);
  float*  xr     = (float*)p;  p += AL((size_t)N * 512 * 4);
  float*  logits = (float*)p;  p += AL((size_t)Etot * HEADS * 4);
  float*  maxv   = (float*)p;  p += AL((size_t)N * HEADS * 4);
  float*  denom  = (float*)p;  p += AL((size_t)N * HEADS * 4);
  float*  outacc = (float*)p;  p += AL((size_t)N * 128 * 4);
  bf16_t* hbf    = (bf16_t*)p; p += AL((size_t)N * 128 * 2);
  bf16_t* wlt    = (bf16_t*)p; p += AL((size_t)512 * 128 * 2);
  bf16_t* wrt    = (bf16_t*)p; p += AL((size_t)512 * 128 * 2);

  // layer-0 input features -> bf16
  {
    int n = N * 128;
    gat_f32_to_bf16<<<(n + 255) / 256, 256, 0, stream>>>(x, hbf, n);
  }

  struct L { int wl, bl, wr, br, att, bias, nout, C, concat; };
  const L Ls[3] = {
    { 2,  3,  4,  5,  6,  7, 128,  32, 1},
    { 8,  9, 10, 11, 12, 13, 128,  32, 1},
    {14, 15, 16, 17, 18, 19, 512, 128, 0},
  };

  for (int i = 0; i < 3; ++i) {
    const L& l = Ls[i];
    const int NOUT = l.nout;
    const int wn = 128 * NOUT;
    gat_convert_w<<<(wn + 255) / 256, 256, 0, stream>>>((const float*)d_in[l.wl], wlt, 128, NOUT);
    gat_convert_w<<<(wn + 255) / 256, 256, 0, stream>>>((const float*)d_in[l.wr], wrt, 128, NOUT);

    const int tiles  = (N / 16) * (NOUT / 64);   // 16x64 tiles per wave
    const int blocks = (tiles + 7) / 8;
    gat_gemm_bf16_wmma<<<blocks, 256, 0, stream>>>(hbf, wlt, (const float*)d_in[l.bl], xl, N, NOUT);
    gat_gemm_bf16_wmma<<<blocks, 256, 0, stream>>>(hbf, wrt, (const float*)d_in[l.br], xr, N, NOUT);

    const int nacc = N * 128;
    gat_init_bufs<<<(nacc + 255) / 256, 256, 0, stream>>>(maxv, denom, outacc, N * HEADS, nacc);

    gat_edge_logits_max<<<(Etot + 7) / 8, 256, 0, stream>>>(esrc, edst, E, N, xl, xr,
        (const float*)d_in[l.att], logits, maxv, l.C);
    const int tot4 = Etot * HEADS;
    gat_edge_exp_sum<<<(tot4 + 255) / 256, 256, 0, stream>>>(edst, E, N, logits, maxv, denom);
    gat_edge_scatter<<<(Etot + 7) / 8, 256, 0, stream>>>(esrc, edst, E, N, xl, logits, denom,
        outacc, l.C, l.concat ? 0 : 1);

    const int last = (i == 2);
    gat_epilogue<<<(nacc + 255) / 256, 256, 0, stream>>>(outacc, (const float*)d_in[l.bias],
        last ? (float*)d_out : nullptr, last ? nullptr : hbf, nacc, last ? 0 : 1);
  }
}
